// MainModel_19542101197175
// MI455X (gfx1250) — matrix operand order repaired and checked by reading, not compiled, not run on hardware
//
#include <hip/hip_runtime.h>

#define HID 96
#define ATOMF 64
#define EDGEF 32
#define LSTR 104   // padded bf16 stride per edge row in LDS (16B-aligned, bank-spread)

typedef __attribute__((ext_vector_type(4)))  unsigned int uint4v;
typedef __attribute__((ext_vector_type(4)))  float        float4v;
typedef __attribute__((ext_vector_type(8)))  float        v8f;
typedef __attribute__((ext_vector_type(16))) __bf16       v16bf;

__device__ __forceinline__ unsigned short f2bf(float f) {
  union { float f; unsigned u; } x; x.f = f;
  unsigned r = x.u + 0x7FFFu + ((x.u >> 16) & 1u);   // round-to-nearest-even
  return (unsigned short)(r >> 16);
}

__device__ __forceinline__ uint4v pack8(float4v a, float4v b) {
  uint4v q;
  q[0] = (unsigned)f2bf(a[0]) | ((unsigned)f2bf(a[1]) << 16);
  q[1] = (unsigned)f2bf(a[2]) | ((unsigned)f2bf(a[3]) << 16);
  q[2] = (unsigned)f2bf(b[0]) | ((unsigned)f2bf(b[1]) << 16);
  q[3] = (unsigned)f2bf(b[2]) | ((unsigned)f2bf(b[3]) << 16);
  return q;
}

// wT[m][n][k] = bf16(W_m[k][n]) : transposed so B fragments are K-contiguous
__global__ void convert_w_kernel(const float* __restrict__ W_init,
                                 const float* __restrict__ W_layers,
                                 unsigned short* __restrict__ wT) {
  int idx = blockIdx.x * blockDim.x + threadIdx.x;
  if (idx >= 4 * HID * HID) return;
  int m = idx / (HID * HID);
  int r = idx % (HID * HID);
  int n = r / HID, k = r % HID;
  float v = (m == 0) ? W_init[k * HID + n]
                     : W_layers[(m - 1) * HID * HID + k * HID + n];
  wT[idx] = f2bf(v);
}

__global__ void zero_kernel(float* __restrict__ p, int n) {
  int i = blockIdx.x * blockDim.x + threadIdx.x;
  if (i < n) p[i] = 0.0f;
}

// mail[dst[e]][:] += hidden[e][:]
__global__ void scatter_mail_kernel(const float* __restrict__ hidden,
                                    const int* __restrict__ dst,
                                    float* __restrict__ mail, int E) {
  int t = blockIdx.x * blockDim.x + threadIdx.x;
  if (t >= E * (HID / 4)) return;
  int e = t / (HID / 4);
  int f = (t % (HID / 4)) * 4;
  float4v h = *(const float4v*)(hidden + e * HID + f);
  float* m = mail + dst[e] * HID + f;
  atomicAdd(m + 0, h[0]);
  atomicAdd(m + 1, h[1]);
  atomicAdd(m + 2, h[2]);
  atomicAdd(m + 3, h[3]);
}

// ---- WMMA tile helpers ---------------------------------------------------
// A (16x32 bf16, ISA layout): lane row=lane&15; kh=lane>>4 selects K halves.
// per-lane data = [kt*32 + kh*8 .. +8) and [kt*32 + 16 + kh*8 .. +8)
__device__ __forceinline__ v16bf load_a_frag(const unsigned short* rowbase,
                                             int kt, int lane) {
  const unsigned short* ap = rowbase + kt * 32 + (lane >> 4) * 8;
  union { uint4v q[2]; v16bf v; } A;
  A.q[0] = *(const uint4v*)ap;          // K offsets +0..7
  A.q[1] = *(const uint4v*)(ap + 16);   // K offsets +16..23
  return A.v;
}
// B (32x16 bf16): lane col=lane&15; K at kt*32 + (lane>>4)*8 .. +8 and +16 .. +24
__device__ __forceinline__ v16bf load_b_frag(const unsigned short* wTm,
                                             int nt, int kt, int lane) {
  const unsigned short* bp = wTm + (nt * 16 + (lane & 15)) * HID +
                             kt * 32 + (lane >> 4) * 8;
  union { uint4v q[2]; v16bf v; } B;
  B.q[0] = *(const uint4v*)bp;
  B.q[1] = *(const uint4v*)(bp + 16);
  return B.v;
}

// init = concat(e_input, n_input[src]) @ W_init ; writes hidden and init
template <bool FULL>
__global__ void __launch_bounds__(128)
init_gemm_kernel(const float* __restrict__ n_in, const float* __restrict__ e_in,
                 const int* __restrict__ src, const unsigned short* __restrict__ wT,
                 float* __restrict__ hidden, float* __restrict__ initb,
                 int ebase0, int E) {
  __shared__ __align__(16) unsigned short sA[4][16 * LSTR];
  const int wave = threadIdx.x >> 5, lane = threadIdx.x & 31;
  const int ebase = ebase0 + blockIdx.x * 64 + wave * 16;

  { // stage 16 edges x 96 feats as bf16 (2 lanes per edge, 48 feats each)
    const int eloc = lane >> 1, half = lane & 1;
    const int edge = ebase + eloc;
    unsigned short* dp = &sA[wave][eloc * LSTR + half * 48];
    if (FULL || edge < E) {
      const int s = src[edge];
      const float* erow = e_in + edge * EDGEF;
      const float* nrow = n_in + s * ATOMF;
      #pragma unroll
      for (int c = 0; c < 6; ++c) {
        int f0 = half * 48 + c * 8;  // chunks never straddle the 32-boundary
        const float* sp = (f0 < EDGEF) ? (erow + f0) : (nrow + (f0 - EDGEF));
        float4v a = *(const float4v*)sp;
        float4v b = *(const float4v*)(sp + 4);
        *(uint4v*)(dp + c * 8) = pack8(a, b);
      }
    } else {
      uint4v z = {0u, 0u, 0u, 0u};
      #pragma unroll
      for (int c = 0; c < 6; ++c) *(uint4v*)(dp + c * 8) = z;
    }
  }
  __syncthreads();

  const int col = lane & 15, kh = lane >> 4;
  const unsigned short* rowbase = &sA[wave][col * LSTR];
  // A tile lives in registers, reused across all 6 N-tiles
  v16bf a0 = load_a_frag(rowbase, 0, lane);
  v16bf a1 = load_a_frag(rowbase, 1, lane);
  v16bf a2 = load_a_frag(rowbase, 2, lane);

  const int obase = (ebase + kh * 8) * HID + col;  // i32 offsets: E*96 < 2^31

  // software-pipelined B: fetch nt+1 fragments while nt's WMMAs run
  v16bf b0 = load_b_frag(wT, 0, 0, lane);
  v16bf b1 = load_b_frag(wT, 0, 1, lane);
  v16bf b2 = load_b_frag(wT, 0, 2, lane);

  #pragma unroll
  for (int nt = 0; nt < 6; ++nt) {
    v16bf n0, n1, n2;
    if (nt < 5) {
      n0 = load_b_frag(wT, nt + 1, 0, lane);
      n1 = load_b_frag(wT, nt + 1, 1, lane);
      n2 = load_b_frag(wT, nt + 1, 2, lane);
    }
    v8f acc = {0.f, 0.f, 0.f, 0.f, 0.f, 0.f, 0.f, 0.f};
    acc = __builtin_amdgcn_wmma_f32_16x16x32_bf16(false, a0, false, b0,
                                                  (short)0, acc, false, false);
    acc = __builtin_amdgcn_wmma_f32_16x16x32_bf16(false, a1, false, b1,
                                                  (short)0, acc, false, false);
    acc = __builtin_amdgcn_wmma_f32_16x16x32_bf16(false, a2, false, b2,
                                                  (short)0, acc, false, false);
    #pragma unroll
    for (int r = 0; r < 8; ++r) {
      if (FULL || ebase + kh * 8 + r < E) {
        int o = obase + nt * 16 + r * HID;
        hidden[o] = acc[r];
        initb[o]  = acc[r];
      }
    }
    b0 = n0; b1 = n1; b2 = n2;
  }
}

// hidden = (mail[src] - hidden) @ W + init   (in-place safe: per-edge rows)
template <bool FULL>
__global__ void __launch_bounds__(128)
layer_gemm_kernel(const float* __restrict__ mail, const int* __restrict__ src,
                  const unsigned short* __restrict__ wT,
                  const float* __restrict__ initb, float* __restrict__ hidden,
                  int ebase0, int E) {
  __shared__ __align__(16) unsigned short sA[4][16 * LSTR];
  const int wave = threadIdx.x >> 5, lane = threadIdx.x & 31;
  const int ebase = ebase0 + blockIdx.x * 64 + wave * 16;

  {
    const int eloc = lane >> 1, half = lane & 1;
    const int edge = ebase + eloc;
    unsigned short* dp = &sA[wave][eloc * LSTR + half * 48];
    if (FULL || edge < E) {
      const int s = src[edge];
      const float* mrow = mail + s * HID;
      const float* hrow = hidden + edge * HID;
      #pragma unroll
      for (int c = 0; c < 6; ++c) {
        int f0 = half * 48 + c * 8;
        float4v ma = *(const float4v*)(mrow + f0);
        float4v mb = *(const float4v*)(mrow + f0 + 4);
        float4v ha = *(const float4v*)(hrow + f0);
        float4v hb = *(const float4v*)(hrow + f0 + 4);
        float4v xa, xb;
        #pragma unroll
        for (int i = 0; i < 4; ++i) { xa[i] = ma[i] - ha[i]; xb[i] = mb[i] - hb[i]; }
        *(uint4v*)(dp + c * 8) = pack8(xa, xb);
      }
    } else {
      uint4v z = {0u, 0u, 0u, 0u};
      #pragma unroll
      for (int c = 0; c < 6; ++c) *(uint4v*)(dp + c * 8) = z;
    }
  }
  __syncthreads();

  const int col = lane & 15, kh = lane >> 4;
  const unsigned short* rowbase = &sA[wave][col * LSTR];
  v16bf a0 = load_a_frag(rowbase, 0, lane);
  v16bf a1 = load_a_frag(rowbase, 1, lane);
  v16bf a2 = load_a_frag(rowbase, 2, lane);

  const int obase = (ebase + kh * 8) * HID + col;

  v16bf b0 = load_b_frag(wT, 0, 0, lane);
  v16bf b1 = load_b_frag(wT, 0, 1, lane);
  v16bf b2 = load_b_frag(wT, 0, 2, lane);

  #pragma unroll
  for (int nt = 0; nt < 6; ++nt) {
    v16bf n0, n1, n2;
    if (nt < 5) {
      n0 = load_b_frag(wT, nt + 1, 0, lane);
      n1 = load_b_frag(wT, nt + 1, 1, lane);
      n2 = load_b_frag(wT, nt + 1, 2, lane);
    }
    v8f acc;
    #pragma unroll
    for (int r = 0; r < 8; ++r) {         // C <- init tile
      if (FULL || ebase + kh * 8 + r < E)
        acc[r] = initb[obase + nt * 16 + r * HID];
      else
        acc[r] = 0.f;
    }
    acc = __builtin_amdgcn_wmma_f32_16x16x32_bf16(false, a0, false, b0,
                                                  (short)0, acc, false, false);
    acc = __builtin_amdgcn_wmma_f32_16x16x32_bf16(false, a1, false, b1,
                                                  (short)0, acc, false, false);
    acc = __builtin_amdgcn_wmma_f32_16x16x32_bf16(false, a2, false, b2,
                                                  (short)0, acc, false, false);
    #pragma unroll
    for (int r = 0; r < 8; ++r) {
      if (FULL || ebase + kh * 8 + r < E)
        hidden[obase + nt * 16 + r * HID] = acc[r];
    }
    b0 = n0; b1 = n1; b2 = n2;
  }
}

// column-sum: out[col] += sum over a slab of rows; blockDim == cols
__global__ void colsum_kernel(const float* __restrict__ X, int rows, int cols,
                              int rows_per_block, float* __restrict__ out) {
  int col = threadIdx.x;
  int r0 = blockIdx.x * rows_per_block;
  int r1 = r0 + rows_per_block; if (r1 > rows) r1 = rows;
  float acc = 0.f;
  for (int r = r0; r < r1; ++r) acc += X[r * cols + col];
  atomicAdd(&out[col], acc);
}

// dgl = s160 @ W_out ; result = dot(concat(dgl, mol), W_full)
__global__ void final_kernel(const float* __restrict__ s160,
                             const float* __restrict__ W_out,
                             const float* __restrict__ mol,
                             const float* __restrict__ W_full,
                             float* __restrict__ out) {
  __shared__ float red[320];
  int t = threadIdx.x;
  float val = 0.f;
  if (t < HID) {
    float d = 0.f;
    for (int k = 0; k < ATOMF + HID; ++k) d += s160[k] * W_out[k * HID + t];
    val = d * W_full[t];
  } else if (t < HID + 200) {
    int m = t - HID;
    val = mol[m] * W_full[HID + m];
  }
  red[t] = val;
  __syncthreads();
  if (t == 0) {
    float s = 0.f;
    for (int i = 0; i < 320; ++i) s += red[i];
    *out = s;
  }
}

extern "C" void kernel_launch(void* const* d_in, const int* in_sizes, int n_in,
                              void* d_out, int out_size, void* d_ws, size_t ws_size,
                              hipStream_t stream) {
  const float* n_in_p   = (const float*)d_in[0];
  const float* e_in_p   = (const float*)d_in[1];
  const float* mol      = (const float*)d_in[2];
  const float* W_init   = (const float*)d_in[3];
  const float* W_layers = (const float*)d_in[4];
  const float* W_out    = (const float*)d_in[5];
  const float* W_full   = (const float*)d_in[6];
  const int*   src      = (const int*)d_in[7];
  const int*   dst      = (const int*)d_in[8];

  const int N = in_sizes[0] / ATOMF;   // 50000
  const int E = in_sizes[7];           // 800000

  // workspace layout (wT first: 16B aligned base from hipMalloc)
  unsigned short* wT = (unsigned short*)d_ws;
  float* fbase  = (float*)((char*)d_ws + (size_t)4 * HID * HID * sizeof(unsigned short));
  float* hidden = fbase;                         // E*96
  float* initb  = hidden + (size_t)E * HID;      // E*96
  float* mail   = initb  + (size_t)E * HID;      // N*96
  float* s160   = mail   + (size_t)N * HID;      // 160

  convert_w_kernel<<<(4 * HID * HID + 255) / 256, 256, 0, stream>>>(W_init, W_layers, wT);

  const int gFull = E / 64;            // guard-free blocks (E=800000 -> all of them)
  const int eTail = gFull * 64;        // first edge of the remainder, if any
  const bool hasTail = (eTail < E);

  init_gemm_kernel<true><<<gFull, 128, 0, stream>>>(n_in_p, e_in_p, src, wT,
                                                    hidden, initb, 0, E);
  if (hasTail)
    init_gemm_kernel<false><<<1, 128, 0, stream>>>(n_in_p, e_in_p, src, wT,
                                                   hidden, initb, eTail, E);

  const int mailN = N * HID;
  const int scatT = E * (HID / 4);
  for (int l = 0; l < 3; ++l) {
    zero_kernel<<<(mailN + 255) / 256, 256, 0, stream>>>(mail, mailN);
    scatter_mail_kernel<<<(scatT + 255) / 256, 256, 0, stream>>>(hidden, dst, mail, E);
    const unsigned short* wTl = wT + (size_t)(1 + l) * HID * HID;
    layer_gemm_kernel<true><<<gFull, 128, 0, stream>>>(mail, src, wTl, initb,
                                                       hidden, 0, E);
    if (hasTail)
      layer_gemm_kernel<false><<<1, 128, 0, stream>>>(mail, src, wTl, initb,
                                                      hidden, eTail, E);
  }

  // readout via linearity: dgl = concat(colsum(n_input), colsum_edges(hidden)) @ W_out
  zero_kernel<<<1, 256, 0, stream>>>(s160, 160);
  colsum_kernel<<<(N + 4095) / 4096, ATOMF, 0, stream>>>(n_in_p, N, ATOMF, 4096, s160);
  colsum_kernel<<<(E + 8191) / 8192, HID, 0, stream>>>(hidden, E, HID, 8192, s160 + ATOMF);

  final_kernel<<<1, 320, 0, stream>>>(s160, W_out, mol, W_full, (float*)d_out);
}
